// RNN_decoder_9698036155134
// MI455X (gfx1250) — matrix operand order repaired and checked by reading, not compiled, not run on hardware
//
#include <hip/hip_runtime.h>
#include <math.h>

#define NODES 32
#define BATCH 64
#define DIMX  32
#define NSTEP 8
#define TTOT  16
#define NE    992
#define HD    64
#define CATD  96

typedef __bf16 bf16;
typedef __attribute__((ext_vector_type(16))) __bf16 v16bf;
typedef __attribute__((ext_vector_type(8)))  __bf16 v8bf;
typedef __attribute__((ext_vector_type(8)))  float  v8f;

// ---------------------------------------------------------------------------
// WMMA helpers (CDNA5: D = A(16x32 bf16) * B(32x16 bf16) + C(16x16 f32))
// ---------------------------------------------------------------------------
static __device__ inline v8f wmma32(const v16bf& a, const v16bf& b, const v8f& c) {
  return __builtin_amdgcn_wmma_f32_16x16x32_bf16(false, a, false, b, (short)0, c,
                                                 false, false);
}

static __device__ inline v8f splat8(float v) {
  v8f r;
#pragma unroll
  for (int i = 0; i < 8; ++i) r[i] = v;
  return r;
}

// B fragment: prepacked as ((kb*NB+nb)*32 + lane)*16 + e (works for global or LDS ptr)
static __device__ inline v16bf load_bfrag(const bf16* frag, int kb, int nb, int NB,
                                          int lane) {
  const bf16* p = frag + ((((kb * NB + nb) << 5) + lane) << 4);
  v8bf lo = *(const v8bf*)p;
  v8bf hi = *(const v8bf*)(p + 8);
  v16bf r;
#pragma unroll
  for (int i = 0; i < 8; ++i) { r[i] = lo[i]; r[8 + i] = hi[i]; }
  return r;
}

// A fragment from a row-major bf16 tile staged in LDS.
static __device__ inline v16bf a_frag_lds(const bf16* row, int kb, int lh) {
  v8bf a = *(const v8bf*)(row + kb * 32 + lh * 8);
  v8bf b = *(const v8bf*)(row + kb * 32 + lh * 8 + 16);
  v16bf r;
#pragma unroll
  for (int i = 0; i < 8; ++i) { r[i] = a[i]; r[8 + i] = b[i]; }
  return r;
}

// A fragment straight from a row-major f32 row in global memory (convert to bf16).
static __device__ inline v16bf a_frag_f32(const float* rowp, int koff, int lh) {
  v16bf r;
  const float* p = rowp + koff + lh * 8;
#pragma unroll
  for (int i = 0; i < 8; ++i) r[i] = (bf16)p[i];
  const float* q = rowp + koff + lh * 8 + 16;
#pragma unroll
  for (int i = 0; i < 8; ++i) r[8 + i] = (bf16)q[i];
  return r;
}

// Scatter a C-layout 8-vector (VGPR j -> row j + 8*lh, col nb*16 + lane&15) into LDS tile.
static __device__ inline void stage_rows(bf16* sw, int SW, int nb, int lane, int lh,
                                         const float* c8) {
  int N = nb * 16 + (lane & 15);
#pragma unroll
  for (int j = 0; j < 8; ++j) sw[(j + 8 * lh) * SW + N] = (bf16)c8[j];
}

static __device__ inline float fsig(float x) { return 1.f / (1.f + __expf(-x)); }

// ---------------------------------------------------------------------------
// Prep kernels
// ---------------------------------------------------------------------------
__global__ void prep_wfrag(const float* __restrict__ W, bf16* __restrict__ out,
                           int Kd, int Nd) {
  int tid = blockIdx.x * blockDim.x + threadIdx.x;
  if (tid >= Kd * Nd) return;
  int e = tid & 15;
  int lane = (tid >> 4) & 31;
  int rest = tid >> 9;
  int NB = Nd >> 4;
  int nb = rest % NB;
  int kb = rest / NB;
  int k = kb * 32 + (e >> 3) * 16 + (lane >> 4) * 8 + (e & 7);
  int nn = nb * 16 + (lane & 15);
  out[tid] = (bf16)W[k * Nd + nn];
}

__global__ void build_elist(const int* __restrict__ es, int* __restrict__ elist,
                            int* __restrict__ ecnt) {
  int n = blockIdx.x * blockDim.x + threadIdx.x;
  if (n >= NODES) return;
  int cnt = 0;
  for (int e = 0; e < NE; ++e) {
    if (es[e] == n && cnt < NODES - 1) { elist[n * (NODES - 1) + cnt] = e; ++cnt; }
  }
  ecnt[n] = cnt;
}

__global__ void gather_x(const float* __restrict__ x, float* __restrict__ xstate, int t) {
  int i = blockIdx.x * blockDim.x + threadIdx.x;
  if (i >= NODES * BATCH * DIMX) return;
  int d = i & (DIMX - 1);
  int b = (i >> 5) & (BATCH - 1);
  int n = i >> 11;
  xstate[(n * BATCH + b) * DIMX + d] = x[(((size_t)b * NODES + n) * DIMX + d) * NSTEP + t];
}

// ---------------------------------------------------------------------------
// Edge phase: per-edge-type MLP mixture + edge GRU, h_edge updated in place.
// One wave = one (edge, 16-batch) tile.  grid = NE blocks x 128 threads (4 waves).
// MLP weight fragments are async-staged global->LDS once per workgroup.
// ---------------------------------------------------------------------------
__global__ __launch_bounds__(128) void edge_kernel(
    const float* __restrict__ xstate, const float* __restrict__ z,
    const int* __restrict__ es,
    const bf16* __restrict__ fW1, const bf16* __restrict__ fW2,
    const bf16* __restrict__ fWi, const bf16* __restrict__ fWh,
    const float* __restrict__ b1, const float* __restrict__ b2,
    const float* __restrict__ bi, const float* __restrict__ bh,
    float* __restrict__ h_edge, int use_gru) {
  const int SW = 72;
  __shared__ alignas(16) bf16 sh[4 * 16 * 72];
  __shared__ alignas(16) bf16 wsh[2 * 3 * 4096];   // fW1 (12288) | fW2 (12288)
  int lane = threadIdx.x & 31;
  int wave = threadIdx.x >> 5;
  int tile = blockIdx.x * 4 + wave;      // 0..3967
  int e = tile >> 2;
  int b0 = (tile & 3) << 4;
  int m = lane & 15, lh = lane >> 4, Nl = lane & 15;
  int colN = es[e];
  int rowN = es[NE + e];
  bf16* sw = sh + wave * 16 * SW;
  bf16* srow = sw + m * SW;

  // --- async tensor-path copy of MLP weight fragments into LDS -------------
  // 2 x 24576B, 128 lanes x 16B per pass -> 12 passes per matrix set.
  {
    unsigned base0 = (unsigned)(uintptr_t)(&wsh[0]);
    unsigned base1 = (unsigned)(uintptr_t)(&wsh[12288]);
    unsigned long long g0 = (unsigned long long)(uintptr_t)fW1;
    unsigned long long g1 = (unsigned long long)(uintptr_t)fW2;
    unsigned tb = (unsigned)threadIdx.x * 16u;
#pragma unroll
    for (int it = 0; it < 12; ++it) {
      unsigned off = (unsigned)it * 2048u + tb;
      unsigned l0 = base0 + off;
      unsigned l1 = base1 + off;
      asm volatile("global_load_async_to_lds_b128 %0, %1, %2"
                   :: "v"(l0), "v"(off), "s"(g0) : "memory");
      asm volatile("global_load_async_to_lds_b128 %0, %1, %2"
                   :: "v"(l1), "v"(off), "s"(g1) : "memory");
    }
    asm volatile("s_wait_asynccnt 0x0" ::: "memory");
  }
  __syncthreads();

  __builtin_prefetch(fWi, 0, 0);
  __builtin_prefetch(fWh, 0, 0);

  // msg = cat(x[row], x[col]) -> A fragments (K=64: kb0 from x[row], kb1 from x[col])
  const float* xr = xstate + (rowN * BATCH + b0 + m) * DIMX;
  const float* xc = xstate + (colN * BATCH + b0 + m) * DIMX;
  v16bf amsg0 = a_frag_f32(xr, 0, lh);
  v16bf amsg1 = a_frag_f32(xc, 0, lh);

  float zs[8][3];
#pragma unroll
  for (int j = 0; j < 8; ++j) {
    const float* zp = z + ((size_t)(b0 + j + 8 * lh) * NE + e) * 3;
    zs[j][0] = zp[0] * (1.f / 3.f);
    zs[j][1] = zp[1] * (1.f / 3.f);
    zs[j][2] = zp[2] * (1.f / 3.f);
  }

  v8f macc[4];
#pragma unroll
  for (int nb = 0; nb < 4; ++nb) macc[nb] = splat8(0.f);

#pragma unroll
  for (int k = 0; k < 3; ++k) {
    const bf16* W1f = &wsh[k * 4096];           // LDS-resident fragments
    const bf16* W2f = &wsh[12288 + k * 4096];
    __syncthreads();
#pragma unroll
    for (int nb = 0; nb < 4; ++nb) {
      v8f acc = splat8(b1[k * HD + nb * 16 + Nl]);
      acc = wmma32(amsg0, load_bfrag(W1f, 0, nb, 4, lane), acc);
      acc = wmma32(amsg1, load_bfrag(W1f, 1, nb, 4, lane), acc);
      float c8[8];
#pragma unroll
      for (int j = 0; j < 8; ++j) c8[j] = fmaxf(acc[j], 0.f);
      stage_rows(sw, SW, nb, lane, lh, c8);
    }
    __syncthreads();
    v16bf a10 = a_frag_lds(srow, 0, lh);
    v16bf a11 = a_frag_lds(srow, 1, lh);
#pragma unroll
    for (int nb = 0; nb < 4; ++nb) {
      v8f acc = splat8(b2[k * HD + nb * 16 + Nl]);
      acc = wmma32(a10, load_bfrag(W2f, 0, nb, 4, lane), acc);
      acc = wmma32(a11, load_bfrag(W2f, 1, nb, 4, lane), acc);
#pragma unroll
      for (int j = 0; j < 8; ++j) macc[nb][j] += fmaxf(acc[j], 0.f) * zs[j][k];
    }
  }

  if (use_gru) {
    __syncthreads();
#pragma unroll
    for (int nb = 0; nb < 4; ++nb) {
      float c8[8];
#pragma unroll
      for (int j = 0; j < 8; ++j) c8[j] = macc[nb][j];
      stage_rows(sw, SW, nb, lane, lh, c8);
    }
    __syncthreads();
    v16bf ai0 = a_frag_lds(srow, 0, lh);
    v16bf ai1 = a_frag_lds(srow, 1, lh);
    const float* hp = h_edge + ((size_t)e * BATCH + b0 + m) * HD;
    v16bf ah0 = a_frag_f32(hp, 0, lh);
    v16bf ah1 = a_frag_f32(hp, 32, lh);
#pragma unroll
    for (int nb = 0; nb < 4; ++nb) {
      int N = nb * 16 + Nl;
      v8f racc = splat8(bi[0 * HD + N] + bh[0 * HD + N]);
      racc = wmma32(ai0, load_bfrag(fWi + 0 * 4096, 0, nb, 4, lane), racc);
      racc = wmma32(ai1, load_bfrag(fWi + 0 * 4096, 1, nb, 4, lane), racc);
      racc = wmma32(ah0, load_bfrag(fWh + 0 * 4096, 0, nb, 4, lane), racc);
      racc = wmma32(ah1, load_bfrag(fWh + 0 * 4096, 1, nb, 4, lane), racc);
      v8f iacc = splat8(bi[1 * HD + N] + bh[1 * HD + N]);
      iacc = wmma32(ai0, load_bfrag(fWi + 1 * 4096, 0, nb, 4, lane), iacc);
      iacc = wmma32(ai1, load_bfrag(fWi + 1 * 4096, 1, nb, 4, lane), iacc);
      iacc = wmma32(ah0, load_bfrag(fWh + 1 * 4096, 0, nb, 4, lane), iacc);
      iacc = wmma32(ah1, load_bfrag(fWh + 1 * 4096, 1, nb, 4, lane), iacc);
      v8f nI = splat8(bi[2 * HD + N]);
      nI = wmma32(ai0, load_bfrag(fWi + 2 * 4096, 0, nb, 4, lane), nI);
      nI = wmma32(ai1, load_bfrag(fWi + 2 * 4096, 1, nb, 4, lane), nI);
      v8f nH = splat8(bh[2 * HD + N]);
      nH = wmma32(ah0, load_bfrag(fWh + 2 * 4096, 0, nb, 4, lane), nH);
      nH = wmma32(ah1, load_bfrag(fWh + 2 * 4096, 1, nb, 4, lane), nH);
#pragma unroll
      for (int j = 0; j < 8; ++j) {
        float rr = fsig(racc[j]);
        float ii = fsig(iacc[j]);
        float nn = tanhf(nI[j] + rr * nH[j]);
        float hid = h_edge[((size_t)e * BATCH + b0 + j + 8 * lh) * HD + N];
        macc[nb][j] = (1.f - ii) * nn + ii * hid;
      }
    }
  }

  // msgs -> h_edge (in place; each row owned by exactly one wave)
#pragma unroll
  for (int nb = 0; nb < 4; ++nb) {
    int N = nb * 16 + Nl;
#pragma unroll
    for (int j = 0; j < 8; ++j)
      h_edge[((size_t)e * BATCH + b0 + j + 8 * lh) * HD + N] = macc[nb][j];
  }
}

// ---------------------------------------------------------------------------
// Scatter-mean to nodes (deterministic edge list order)
// ---------------------------------------------------------------------------
__global__ void agg_kernel(const float* __restrict__ msgs, const int* __restrict__ elist,
                           const int* __restrict__ ecnt, float* __restrict__ agg) {
  int tix = blockIdx.x * blockDim.x + threadIdx.x;
  if (tix >= NODES * BATCH * HD) return;
  int o = tix & (HD - 1);
  int b = (tix >> 6) & (BATCH - 1);
  int n = tix >> 12;
  int cnt = ecnt[n];
  float s = 0.f;
  for (int j = 0; j < cnt; ++j) {
    int e = elist[n * (NODES - 1) + j];
    s += msgs[((size_t)e * BATCH + b) * HD + o];
  }
  agg[tix] = s / (float)(cnt > 0 ? cnt : 1);
}

// ---------------------------------------------------------------------------
// Node phase: node GRU (96-dim) + 3-layer output MLP; xstate & h_node in place.
// grid = NODES blocks x 128 threads (4 waves; one wave per 16-batch tile).
// ---------------------------------------------------------------------------
__global__ __launch_bounds__(128) void node_kernel(
    float* __restrict__ xstate, const float* __restrict__ agg,
    const bf16* __restrict__ fWi, const bf16* __restrict__ fWh,
    const bf16* __restrict__ fO1, const bf16* __restrict__ fO2,
    const bf16* __restrict__ fO3,
    const float* __restrict__ bi, const float* __restrict__ bh,
    const float* __restrict__ ob1, const float* __restrict__ ob2,
    const float* __restrict__ ob3,
    float* __restrict__ h_node, float* __restrict__ out, int t, int use_gru) {
  const int SW = 104;
  __shared__ alignas(16) bf16 sh[4 * 16 * 104];
  int lane = threadIdx.x & 31;
  int wave = threadIdx.x >> 5;
  int tile = blockIdx.x * 4 + wave;   // 0..127
  int n = tile >> 2;
  int b0 = (tile & 3) << 4;
  int m = lane & 15, lh = lane >> 4, Nl = lane & 15;
  bf16* sw = sh + wave * 16 * SW;
  bf16* srow = sw + m * SW;

  __builtin_prefetch(fWi, 0, 0);
  __builtin_prefetch(fO1, 0, 0);

  // cat = [x_in (32), agg (64)]  -> A fragments (K = 96)
  const float* xp = xstate + (n * BATCH + b0 + m) * DIMX;
  const float* ap = agg + (n * BATCH + b0 + m) * HD;
  v16bf acat[3];
  acat[0] = a_frag_f32(xp, 0, lh);
  acat[1] = a_frag_f32(ap, 0, lh);
  acat[2] = a_frag_f32(ap, 32, lh);

  float catC[6][8];
#pragma unroll
  for (int nb = 0; nb < 6; ++nb) {
    int N = nb * 16 + Nl;
#pragma unroll
    for (int j = 0; j < 8; ++j) {
      int bb = b0 + j + 8 * lh;
      catC[nb][j] = (N < DIMX) ? xstate[(n * BATCH + bb) * DIMX + N]
                               : agg[(n * BATCH + bb) * HD + (N - DIMX)];
    }
  }

  if (use_gru) {
    const float* hp = h_node + (n * BATCH + b0 + m) * CATD;
    v16bf ah[3];
    ah[0] = a_frag_f32(hp, 0, lh);
    ah[1] = a_frag_f32(hp, 32, lh);
    ah[2] = a_frag_f32(hp, 64, lh);
#pragma unroll
    for (int nb = 0; nb < 6; ++nb) {
      int N = nb * 16 + Nl;
      v8f racc = splat8(bi[0 * CATD + N] + bh[0 * CATD + N]);
      v8f iacc = splat8(bi[1 * CATD + N] + bh[1 * CATD + N]);
      v8f nI = splat8(bi[2 * CATD + N]);
      v8f nH = splat8(bh[2 * CATD + N]);
#pragma unroll
      for (int kb = 0; kb < 3; ++kb) {
        racc = wmma32(acat[kb], load_bfrag(fWi + 0 * 9216, kb, nb, 6, lane), racc);
        racc = wmma32(ah[kb],   load_bfrag(fWh + 0 * 9216, kb, nb, 6, lane), racc);
        iacc = wmma32(acat[kb], load_bfrag(fWi + 1 * 9216, kb, nb, 6, lane), iacc);
        iacc = wmma32(ah[kb],   load_bfrag(fWh + 1 * 9216, kb, nb, 6, lane), iacc);
        nI   = wmma32(acat[kb], load_bfrag(fWi + 2 * 9216, kb, nb, 6, lane), nI);
        nH   = wmma32(ah[kb],   load_bfrag(fWh + 2 * 9216, kb, nb, 6, lane), nH);
      }
#pragma unroll
      for (int j = 0; j < 8; ++j) {
        float rr = fsig(racc[j]);
        float ii = fsig(iacc[j]);
        float nn = tanhf(nI[j] + rr * nH[j]);
        float hid = h_node[(n * BATCH + b0 + j + 8 * lh) * CATD + N];
        catC[nb][j] = (1.f - ii) * nn + ii * hid;
      }
    }
  }

  // persist h_node = cat (post-GRU)
#pragma unroll
  for (int nb = 0; nb < 6; ++nb) {
    int N = nb * 16 + Nl;
#pragma unroll
    for (int j = 0; j < 8; ++j)
      h_node[(n * BATCH + b0 + j + 8 * lh) * CATD + N] = catC[nb][j];
  }

  // output MLP: 96 -> 64 -> 64 -> 32
  __syncthreads();
#pragma unroll
  for (int nb = 0; nb < 6; ++nb) stage_rows(sw, SW, nb, lane, lh, catC[nb]);
  __syncthreads();
  v16bf ac[3];
#pragma unroll
  for (int kb = 0; kb < 3; ++kb) ac[kb] = a_frag_lds(srow, kb, lh);
  float h1s[4][8];
#pragma unroll
  for (int nb = 0; nb < 4; ++nb) {
    v8f acc = splat8(ob1[nb * 16 + Nl]);
#pragma unroll
    for (int kb = 0; kb < 3; ++kb)
      acc = wmma32(ac[kb], load_bfrag(fO1, kb, nb, 4, lane), acc);
#pragma unroll
    for (int j = 0; j < 8; ++j) h1s[nb][j] = fmaxf(acc[j], 0.f);
  }
  __syncthreads();
#pragma unroll
  for (int nb = 0; nb < 4; ++nb) stage_rows(sw, SW, nb, lane, lh, h1s[nb]);
  __syncthreads();
  v16bf a20 = a_frag_lds(srow, 0, lh);
  v16bf a21 = a_frag_lds(srow, 1, lh);
  float h2s[4][8];
#pragma unroll
  for (int nb = 0; nb < 4; ++nb) {
    v8f acc = splat8(ob2[nb * 16 + Nl]);
    acc = wmma32(a20, load_bfrag(fO2, 0, nb, 4, lane), acc);
    acc = wmma32(a21, load_bfrag(fO2, 1, nb, 4, lane), acc);
#pragma unroll
    for (int j = 0; j < 8; ++j) h2s[nb][j] = fmaxf(acc[j], 0.f);
  }
  __syncthreads();
#pragma unroll
  for (int nb = 0; nb < 4; ++nb) stage_rows(sw, SW, nb, lane, lh, h2s[nb]);
  __syncthreads();
  v16bf a30 = a_frag_lds(srow, 0, lh);
  v16bf a31 = a_frag_lds(srow, 1, lh);
#pragma unroll
  for (int nb = 0; nb < 2; ++nb) {
    v8f dacc = splat8(ob3[nb * 16 + Nl]);
    dacc = wmma32(a30, load_bfrag(fO3, 0, nb, 2, lane), dacc);
    dacc = wmma32(a31, load_bfrag(fO3, 1, nb, 2, lane), dacc);
    int N = nb * 16 + Nl;
#pragma unroll
    for (int j = 0; j < 8; ++j) {
      int bb = b0 + j + 8 * lh;
      float xm = xstate[(n * BATCH + bb) * DIMX + N] + dacc[j];
      xstate[(n * BATCH + bb) * DIMX + N] = xm;
      size_t base = ((size_t)(bb * NODES + n) * DIMX + N);
      out[524288 + base * TTOT + t] = xm;        // x_hat (output 1)
      out[1572864 + base * TTOT + t] = xm;       // x_hat (output 2)
      if (t >= NSTEP) out[base * NSTEP + (t - NSTEP)] = xm;  // last PRED_LEN steps
    }
  }
}

// ---------------------------------------------------------------------------
// Host orchestration
// ---------------------------------------------------------------------------
extern "C" void kernel_launch(void* const* d_in, const int* in_sizes, int n_in,
                              void* d_out, int out_size, void* d_ws, size_t ws_size,
                              hipStream_t stream) {
  (void)in_sizes; (void)n_in; (void)out_size; (void)ws_size;
  const float* xin   = (const float*)d_in[0];
  const float* z     = (const float*)d_in[1];
  const int*   es    = (const int*)d_in[2];
  const float* msgW1 = (const float*)d_in[6];
  const float* msgb1 = (const float*)d_in[7];
  const float* msgW2 = (const float*)d_in[8];
  const float* msgb2 = (const float*)d_in[9];
  const float* oW1   = (const float*)d_in[10];
  const float* ob1   = (const float*)d_in[11];
  const float* oW2   = (const float*)d_in[12];
  const float* ob2   = (const float*)d_in[13];
  const float* oW3   = (const float*)d_in[14];
  const float* ob3   = (const float*)d_in[15];
  const float* geWi  = (const float*)d_in[16];
  const float* gebi  = (const float*)d_in[17];
  const float* geWh  = (const float*)d_in[18];
  const float* gebh  = (const float*)d_in[19];
  const float* gnWi  = (const float*)d_in[20];
  const float* gnbi  = (const float*)d_in[21];
  const float* gnWh  = (const float*)d_in[22];
  const float* gnbh  = (const float*)d_in[23];

  size_t off = 0;
  auto walloc = [&](size_t bytes) -> char* {
    off = (off + 255) & ~(size_t)255;
    char* p = (char*)d_ws + off;
    off += bytes;
    return p;
  };
  float* xstate = (float*)walloc((size_t)NODES * BATCH * DIMX * sizeof(float));
  float* h_edge = (float*)walloc((size_t)NE * BATCH * HD * sizeof(float));
  float* aggb   = (float*)walloc((size_t)NODES * BATCH * HD * sizeof(float));
  float* h_node = (float*)walloc((size_t)NODES * BATCH * CATD * sizeof(float));
  int*   ecnt   = (int*)walloc(NODES * sizeof(int));
  int*   elist  = (int*)walloc((size_t)NODES * (NODES - 1) * sizeof(int));
  bf16*  fW1    = (bf16*)walloc((size_t)3 * 4096 * sizeof(bf16));
  bf16*  fW2    = (bf16*)walloc((size_t)3 * 4096 * sizeof(bf16));
  bf16*  fgeWi  = (bf16*)walloc((size_t)3 * 4096 * sizeof(bf16));
  bf16*  fgeWh  = (bf16*)walloc((size_t)3 * 4096 * sizeof(bf16));
  bf16*  fgnWi  = (bf16*)walloc((size_t)3 * 9216 * sizeof(bf16));
  bf16*  fgnWh  = (bf16*)walloc((size_t)3 * 9216 * sizeof(bf16));
  bf16*  fO1    = (bf16*)walloc((size_t)6144 * sizeof(bf16));
  bf16*  fO2    = (bf16*)walloc((size_t)4096 * sizeof(bf16));
  bf16*  fO3    = (bf16*)walloc((size_t)2048 * sizeof(bf16));

  build_elist<<<1, 32, 0, stream>>>(es, elist, ecnt);
  for (int k = 0; k < 3; ++k) {
    prep_wfrag<<<16, 256, 0, stream>>>(msgW1 + k * 4096, fW1 + k * 4096, 64, 64);
    prep_wfrag<<<16, 256, 0, stream>>>(msgW2 + k * 4096, fW2 + k * 4096, 64, 64);
    prep_wfrag<<<16, 256, 0, stream>>>(geWi + k * 4096, fgeWi + k * 4096, 64, 64);
    prep_wfrag<<<16, 256, 0, stream>>>(geWh + k * 4096, fgeWh + k * 4096, 64, 64);
    prep_wfrag<<<36, 256, 0, stream>>>(gnWi + k * 9216, fgnWi + k * 9216, 96, 96);
    prep_wfrag<<<36, 256, 0, stream>>>(gnWh + k * 9216, fgnWh + k * 9216, 96, 96);
  }
  prep_wfrag<<<24, 256, 0, stream>>>(oW1, fO1, 96, 64);
  prep_wfrag<<<16, 256, 0, stream>>>(oW2, fO2, 64, 64);
  prep_wfrag<<<8, 256, 0, stream>>>(oW3, fO3, 64, 32);

  for (int t = 0; t < TTOT; ++t) {
    if (t < NSTEP)
      gather_x<<<(NODES * BATCH * DIMX + 255) / 256, 256, 0, stream>>>(xin, xstate, t);
    edge_kernel<<<NE, 128, 0, stream>>>(xstate, z, es, fW1, fW2, fgeWi, fgeWh,
                                        msgb1, msgb2, gebi, gebh, h_edge,
                                        t > 0 ? 1 : 0);
    agg_kernel<<<(NODES * BATCH * HD + 255) / 256, 256, 0, stream>>>(h_edge, elist,
                                                                     ecnt, aggb);
    node_kernel<<<NODES, 128, 0, stream>>>(xstate, aggb, fgnWi, fgnWh, fO1, fO2, fO3,
                                           gnbi, gnbh, ob1, ob2, ob3, h_node,
                                           (float*)d_out, t, t > 0 ? 1 : 0);
  }
}
